// FeedForward_80848464380126
// MI455X (gfx1250) — 8-bit matrix operand order repaired and checked by reading, compiled, not run on hardware
//
#include <hip/hip_runtime.h>
#include <stdint.h>

typedef __attribute__((ext_vector_type(8))) int          v8i;
typedef __attribute__((ext_vector_type(4))) int          v4i;
typedef __attribute__((ext_vector_type(4))) unsigned int v4u;

#define EPSV 1e-5f

#define BM 128
#define BN 256
#define BK 64

#define LDS_A_BYTES (BM * BK)
#define LDS_B_BYTES (BN * BK)
#define LDS_B_BASE  (2 * LDS_A_BYTES)
#define LDS_TOTAL   (2 * LDS_A_BYTES + 2 * LDS_B_BYTES)

__device__ __forceinline__ void tdm_load_tile_2d(
    uint32_t lds_off, const void* gptr,
    uint32_t tile_w_bytes, uint32_t tile_rows,
    uint64_t row_stride_bytes, uint64_t tensor_d0, uint64_t tensor_d1)
{
  uint64_t ga = (uint64_t)(uintptr_t)gptr;
  v4u g0;
  g0.x = 1u;
  g0.y = lds_off;
  g0.z = (uint32_t)ga;
  g0.w = (uint32_t)((ga >> 32) & 0x01FFFFFFull) | (2u << 30);
  v8i g1;
  g1[0] = 0;
  g1[1] = (int)((tensor_d0 & 0xFFFFull) << 16);
  g1[2] = (int)(((tensor_d0 >> 16) & 0xFFFFull) |
                ((tensor_d1 & 0xFFFFull) << 16));
  g1[3] = (int)(((tensor_d1 >> 16) & 0xFFFFull) |
                ((uint64_t)(tile_w_bytes & 0xFFFFu) << 16));
  g1[4] = (int)(tile_rows & 0xFFFFu);
  g1[5] = (int)(row_stride_bytes & 0xFFFFFFFFull);
  g1[6] = (int)((row_stride_bytes >> 32) & 0xFFFFull);
  g1[7] = 0;
  v4i g2 = {0, 0, 0, 0};
  v4i g3 = {0, 0, 0, 0};
#if __clang_major__ >= 23
  v8i g4 = {0, 0, 0, 0, 0, 0, 0, 0};
  __builtin_amdgcn_tensor_load_to_lds(g0, g1, g2, g3, g4, 0);
#else
  __builtin_amdgcn_tensor_load_to_lds(g0, g1, g2, g3, 0);
#endif
}

__device__ __forceinline__ v8i ldsA_frag(const signed char* rowp, int koff) {
  const int* a = (const int*)(rowp + koff);
  const int* b = (const int*)(rowp + koff + 16);
  const int* c = (const int*)(rowp + koff + 32);
  const int* d = (const int*)(rowp + koff + 48);
  v8i f;
  f[0] = a[0]; f[1] = a[1]; f[2] = b[0]; f[3] = b[1];
  f[4] = c[0]; f[5] = c[1]; f[6] = d[0]; f[7] = d[1];
  return f;
}
__device__ __forceinline__ v8i ldsB_frag(const signed char* rowp, int koff) {
  const int* a = (const int*)(rowp + koff);
  const int* b = (const int*)(rowp + koff + 16);
  const int* c = (const int*)(rowp + koff + 32);
  const int* d = (const int*)(rowp + koff + 48);
  v8i f;
  f[0] = a[0]; f[1] = a[1]; f[2] = b[0]; f[3] = b[1];
  f[4] = c[0]; f[5] = c[1]; f[6] = d[0]; f[7] = d[1];
  return f;
}

template <bool RELU2>
__global__ __launch_bounds__(256) void bitgemm_i8(
    const signed char* __restrict__ A,
    const signed char* __restrict__ B,
    float* __restrict__ C,
    const float* __restrict__ arow_scale,
    const float* __restrict__ wscale_p,
    int M, int N, int K)
{
  __shared__ __align__(16) signed char sm[LDS_TOTAL];

  const int tid  = threadIdx.x;
  const int lane = tid & 31;
  const int wave = tid >> 5;
  const int wm   = (wave & 1) * 64;
  const int wn   = (wave >> 1) * 64;

  const int m0 = blockIdx.y * BM;
  const int n0 = blockIdx.x * BN;

  const signed char* Ap = A + (size_t)m0 * K;
  const signed char* Bp = B + (size_t)n0 * K;

  v8i acc[4][4];
#pragma unroll
  for (int i = 0; i < 4; ++i)
#pragma unroll
    for (int j = 0; j < 4; ++j)
      acc[i][j] = (v8i){0, 0, 0, 0, 0, 0, 0, 0};

  const int lrow  = lane & 15;
  const int koffA = (lane >> 4) * 8;
  const int koffB = (lane >> 4) * 8;

  if (wave == 0) {
    tdm_load_tile_2d(0u, Ap, BK, BM, (uint64_t)K, (uint64_t)K, (uint64_t)M);
    tdm_load_tile_2d(LDS_B_BASE, Bp, BK, BN, (uint64_t)K, (uint64_t)K, (uint64_t)N);
  }
  __builtin_amdgcn_s_wait_tensorcnt(0);
  __syncthreads();

  int cur = 0;
  for (int kt = 0; kt < K; kt += BK) {
    if (wave == 0 && kt + BK < K) {
      const int nxt = cur ^ 1;
      tdm_load_tile_2d((uint32_t)(nxt * LDS_A_BYTES), Ap + kt + BK,
                       BK, BM, (uint64_t)K, (uint64_t)K, (uint64_t)M);
      tdm_load_tile_2d((uint32_t)(LDS_B_BASE + nxt * LDS_B_BYTES), Bp + kt + BK,
                       BK, BN, (uint64_t)K, (uint64_t)K, (uint64_t)N);
    }

    const signed char* sA = sm + cur * LDS_A_BYTES;
    const signed char* sB = sm + LDS_B_BASE + cur * LDS_B_BYTES;

    v8i af[4], bf[4];
#pragma unroll
    for (int i = 0; i < 4; ++i)
      af[i] = ldsA_frag(sA + (wm + i * 16 + lrow) * BK, koffA);
#pragma unroll
    for (int j = 0; j < 4; ++j)
      bf[j] = ldsB_frag(sB + (wn + j * 16 + lrow) * BK, koffB);

#pragma unroll
    for (int i = 0; i < 4; ++i)
#pragma unroll
      for (int j = 0; j < 4; ++j)
        acc[i][j] = __builtin_amdgcn_wmma_i32_16x16x64_iu8(
            true, af[i], true, bf[j], acc[i][j], false, false);

    __syncthreads();
    __builtin_amdgcn_s_wait_tensorcnt(0);
    __syncthreads();
    cur ^= 1;
  }

  const float wsc   = *wscale_p;
  const int   mhalf = (lane >> 4) * 8;
  const int   ncol  = lane & 15;
#pragma unroll
  for (int i = 0; i < 4; ++i) {
    const int gmb = m0 + wm + i * 16 + mhalf;
#pragma unroll
    for (int j = 0; j < 4; ++j) {
      const int gn = n0 + wn + j * 16 + ncol;
#pragma unroll
      for (int r = 0; r < 8; ++r) {
        const int gm = gmb + r;
        float y = (float)acc[i][j][r] * arow_scale[gm] * wsc;
        if (RELU2) { y = fmaxf(y, 0.0f); y = y * y; }
        C[(size_t)gm * N + gn] = y;
      }
    }
  }
}

__global__ void zero2_kernel(float* a, float* b) {
  if (threadIdx.x == 0 && blockIdx.x == 0) { *a = 0.f; *b = 0.f; }
}

__global__ __launch_bounds__(256) void absclip_sum_kernel(
    const float* __restrict__ W, float* __restrict__ acc, size_t n)
{
  float s = 0.f;
  for (size_t i = (size_t)blockIdx.x * blockDim.x + threadIdx.x; i < n;
       i += (size_t)gridDim.x * blockDim.x) {
    float w = fminf(fmaxf(W[i], -1.0f), 1.0f);
    s += fabsf(w);
  }
#pragma unroll
  for (int off = 16; off > 0; off >>= 1) s += __shfl_down(s, off, 32);
  __shared__ float red[8];
  if ((threadIdx.x & 31) == 0) red[threadIdx.x >> 5] = s;
  __syncthreads();
  if (threadIdx.x == 0) {
    float t = 0.f;
    for (int i = 0; i < 8; ++i) t += red[i];
    atomicAdd(acc, t);
  }
}

__global__ void wscale_fin_kernel(const float* su, const float* sd,
                                  float* scu, float* scd,
                                  float inv_nu, float inv_nd) {
  if (threadIdx.x == 0 && blockIdx.x == 0) {
    *scu = (*su) * inv_nu + EPSV;
    *scd = (*sd) * inv_nd + EPSV;
  }
}

__global__ __launch_bounds__(256) void weight_quant_kernel(
    const float* __restrict__ W, signed char* __restrict__ Q,
    const float* __restrict__ scale_p, size_t n)
{
  const float inv = 1.0f / (*scale_p);
  for (size_t i = (size_t)blockIdx.x * blockDim.x + threadIdx.x; i < n;
       i += (size_t)gridDim.x * blockDim.x) {
    float w = fminf(fmaxf(W[i], -1.0f), 1.0f);
    float q = rintf(w * inv);
    q = fminf(fmaxf(q, -1.0f), 1.0f);
    Q[i] = (signed char)(int)q;
  }
}

__global__ __launch_bounds__(256) void act_quant_rows_kernel(
    const float* __restrict__ X, signed char* __restrict__ Q,
    float* __restrict__ rscale, int rowlen)
{
  const int row = blockIdx.x;
  const float* xr = X + (size_t)row * rowlen;
  float amax = 0.f;
  for (int i = threadIdx.x; i < rowlen; i += blockDim.x)
    amax = fmaxf(amax, fabsf(xr[i]));
#pragma unroll
  for (int off = 16; off > 0; off >>= 1)
    amax = fmaxf(amax, __shfl_down(amax, off, 32));
  __shared__ float red[8];
  if ((threadIdx.x & 31) == 0) red[threadIdx.x >> 5] = amax;
  __syncthreads();
  if (threadIdx.x == 0) {
    float m = red[0];
    for (int i = 1; i < 8; ++i) m = fmaxf(m, red[i]);
    red[0] = fmaxf(m, EPSV);
  }
  __syncthreads();
  const float am = red[0];
  const float s  = 127.0f / am;
  if (threadIdx.x == 0) rscale[row] = am * (1.0f / 127.0f);
  signed char* qr = Q + (size_t)row * rowlen;
  for (int i = threadIdx.x; i < rowlen; i += blockDim.x) {
    float v = rintf(xr[i] * s);
    v = fminf(fmaxf(v, -128.0f), 127.0f);
    qr[i] = (signed char)(int)v;
  }
}

extern "C" void kernel_launch(void* const* d_in, const int* in_sizes, int n_in,
                              void* d_out, int out_size, void* d_ws, size_t ws_size,
                              hipStream_t stream)
{
  (void)n_in; (void)out_size; (void)ws_size;
  const float* x    = (const float*)d_in[0];
  const float* w_up = (const float*)d_in[1];
  const float* w_dn = (const float*)d_in[2];
  float* out = (float*)d_out;

  const int H = 4096, F = 16384;
  const int T = in_sizes[0] / H;

  char*  ws       = (char*)d_ws;
  float* wsum_up  = (float*)(ws + 0);
  float* wsum_dn  = (float*)(ws + 4);
  float* wsc_up   = (float*)(ws + 8);
  float* wsc_dn   = (float*)(ws + 12);
  float* ascale_x = (float*)(ws + 256);
  float* ascale_h = (float*)(ws + 256 + (size_t)T * 4);
  size_t off = 256 + (size_t)2 * T * 4;
  off = (off + 255) & ~(size_t)255;
  signed char* xq    = (signed char*)(ws + off); off += (size_t)T * H;
  signed char* hq    = (signed char*)(ws + off); off += (size_t)T * F;
  signed char* wq_up = (signed char*)(ws + off); off += (size_t)F * H;
  signed char* wq_dn = (signed char*)(ws + off); off += (size_t)H * F;
  off = (off + 255) & ~(size_t)255;
  float* h = (float*)(ws + off);

  zero2_kernel<<<1, 1, 0, stream>>>(wsum_up, wsum_dn);
  absclip_sum_kernel<<<4096, 256, 0, stream>>>(w_up, wsum_up, (size_t)F * H);
  absclip_sum_kernel<<<4096, 256, 0, stream>>>(w_dn, wsum_dn, (size_t)H * F);
  wscale_fin_kernel<<<1, 1, 0, stream>>>(wsum_up, wsum_dn, wsc_up, wsc_dn,
                                         1.0f / ((float)F * (float)H),
                                         1.0f / ((float)H * (float)F));
  weight_quant_kernel<<<8192, 256, 0, stream>>>(w_up, wq_up, wsc_up, (size_t)F * H);
  weight_quant_kernel<<<8192, 256, 0, stream>>>(w_dn, wq_dn, wsc_dn, (size_t)H * F);
  act_quant_rows_kernel<<<T, 256, 0, stream>>>(x, xq, ascale_x, H);
  bitgemm_i8<true><<<dim3(F / BN, T / BM), 256, 0, stream>>>(
      xq, wq_up, h, ascale_x, wsc_up, T, F, H);
  act_quant_rows_kernel<<<T, 256, 0, stream>>>(h, hq, ascale_h, F);
  bitgemm_i8<false><<<dim3(H / BN, T / BM), 256, 0, stream>>>(
      hq, wq_dn, out, ascale_h, wsc_dn, T, H, F);
}
